// OptionNet_85976655331415
// MI455X (gfx1250) — matrix operand order repaired and checked by reading, not compiled, not run on hardware
//
#include <hip/hip_runtime.h>
#include <hip/hip_bf16.h>
#include <stdint.h>

// ---------------- problem constants ----------------
#define NB 32768   // agents
#define DD 512     // obs dim
#define HH 1024    // hidden
#define NE 10      // experts: 0=meta, 1=term, 2..9=options
#define NKT (DD/32)   // 16 K-tiles of 32 for GEMM1
#define NHT (HH/16)   // 64 H tiles of 16
#define NHK (HH/32)   // 32 K-tiles of 32 for head GEMM

typedef __attribute__((ext_vector_type(16))) __bf16 v16bf;
typedef __attribute__((ext_vector_type(8)))  float  v8f;

static __device__ __forceinline__ unsigned short f2bf(float f) {
  unsigned u = __builtin_bit_cast(unsigned, f);
  u += 0x7FFFu + ((u >> 16) & 1u);   // round-to-nearest-even
  return (unsigned short)(u >> 16);
}

// ---------------- pre-pass: obs f32 -> bf16 row-major ----------------
__global__ __launch_bounds__(256) void k_conv_obs(const float* __restrict__ src,
                                                  unsigned short* __restrict__ dst, int n) {
  int i = blockIdx.x * 256 + threadIdx.x;
  if (i < n) dst[i] = f2bf(src[i]);
}

// ---------------- pre-pass: GEMM1 weights -> B-matrix lane layout ----------------
// dst u32 index = tileIdx*256 + lane*8 + pair, tileIdx = (e*NHT + hTile)*NKT + kTile
// lane: n = hTile*16 + lane%16; k = kTile*32 + (lane<16?0:16) + 2*pair
__global__ __launch_bounds__(256) void k_swz_w(const float* __restrict__ Wm1,
                                               const float* __restrict__ Wt1,
                                               const float* __restrict__ W1,
                                               unsigned int* __restrict__ dst) {
  int idx = blockIdx.x * 256 + threadIdx.x;
  if (idx >= NE * NHT * NKT * 32 * 8) return;
  int pair = idx & 7, lane = (idx >> 3) & 31, kt = (idx >> 8) & 15, ht = (idx >> 12) & 63, e = idx >> 18;
  int n = ht * 16 + (lane & 15);
  int k = kt * 32 + ((lane & 16) ? 16 : 0) + pair * 2;
  const float* src = (e == 0) ? Wm1 : (e == 1) ? Wt1 : (W1 + (size_t)(e - 2) * DD * HH);
  unsigned lo = f2bf(src[(size_t)k * HH + n]);
  unsigned hi = f2bf(src[(size_t)(k + 1) * HH + n]);
  dst[idx] = lo | (hi << 16);
}

// ---------------- pre-pass: head weights (padded to 2 N-tiles) -> B layout ----------------
// dst u32 index = ((e*2+nt)*NHK + hk)*256 + lane*8 + pair
// col = nt*16 + lane%16 ; k = hk*32 + (lane<16?0:16) + 2*pair
// meta: cols 0-7 = Wm_pi, col 8 = Wm_v ; term: cols 0-7 = Wt2
// option o: cols 0-15 = Wpi[o], col 16 = Wv[o] ; others 0
__global__ __launch_bounds__(256) void k_swz_heads(const float* __restrict__ Wm_pi,
                                                   const float* __restrict__ Wm_v,
                                                   const float* __restrict__ Wt2,
                                                   const float* __restrict__ Wpi,
                                                   const float* __restrict__ Wv,
                                                   unsigned int* __restrict__ dst) {
  int idx = blockIdx.x * 256 + threadIdx.x;
  if (idx >= NE * 2 * NHK * 32 * 8) return;
  int pair = idx & 7, lane = (idx >> 3) & 31, hk = (idx >> 8) & 31, nt = (idx >> 13) & 1, e = idx >> 14;
  int col = nt * 16 + (lane & 15);
  int k = hk * 32 + ((lane & 16) ? 16 : 0) + pair * 2;
  float v0 = 0.f, v1 = 0.f;
  for (int s = 0; s < 2; ++s) {
    int kk = k + s;
    float val = 0.f;
    if (e == 0)      val = (col < 8) ? Wm_pi[(size_t)kk * 8 + col] : (col == 8 ? Wm_v[kk] : 0.f);
    else if (e == 1) val = (col < 8) ? Wt2[(size_t)kk * 8 + col] : 0.f;
    else {
      int o = e - 2;
      val = (col < 16) ? Wpi[((size_t)o * HH + kk) * 16 + col]
                       : (col == 16 ? Wv[(size_t)o * HH + kk] : 0.f);
    }
    if (s == 0) v0 = val; else v1 = val;
  }
  dst[idx] = (unsigned)f2bf(v0) | ((unsigned)f2bf(v1) << 16);
}

// ---------------- fused option-critic kernel ----------------
// 256 threads = 8 waves. Block owns 64 rows. Wave w: rowgroup rg=w/2 (16 rows), nt=w&1.
// A (obs stripe, 16x512 bf16) is held in 128 VGPRs for the whole kernel.
// Every 4th 1KB k-tile (8KB) of the next chunk's 32KB B slab is prefetched at WGP scope so those
// per-k-tile B loads resolve from the near cache.
__global__ __launch_bounds__(256) void k_option_net(
    const unsigned short* __restrict__ obsb,  // [NB][DD] bf16
    const unsigned int*   __restrict__ wsw,   // swizzled GEMM1 weights
    const unsigned int*   __restrict__ hsw,   // swizzled head weights
    const float* __restrict__ bm1,
    const float* __restrict__ b1,             // [8][HH]
    const unsigned char* __restrict__ dones,
    const int* __restrict__ eopt,
    float* __restrict__ out) {
  __shared__ unsigned short hbuf[64 * 64];    // bf16 relu(h) chunk
  __shared__ float s_meta_act[64], s_meta_lp[64], s_meta_val[64];
  __shared__ float s_term[64 * 8];
  __shared__ float s_oact[64 * 8], s_olp[64 * 8], s_oval[64 * 8];

  const int tid  = threadIdx.x;
  const int lane = tid & 31;
  const int w    = tid >> 5;
  const int rg   = w >> 1;         // 0..3
  const int nt   = w & 1;          // head N-tile / 32-col half
  const int c16  = lane & 15;
  const int hi   = lane >> 4;      // 0: lanes 0-15, 1: lanes 16-31
  const int rowblk = blockIdx.x * 64;

  union AB { v16bf v; uint4 q[2]; };

  // ---- preload A operand once: this lane's obs row (A layout: lane l -> M = l%16) ----
  const unsigned short* aRow = obsb + (size_t)(rowblk + rg * 16 + c16) * DD;
  AB Areg[NKT];
  #pragma unroll
  for (int kt = 0; kt < NKT; ++kt) {
    const unsigned short* ap = aRow + kt * 32 + hi * 16;
    Areg[kt].q[0] = *(const uint4*)(ap);
    Areg[kt].q[1] = *(const uint4*)(ap + 8);
  }

  for (int e = 0; e < NE; ++e) {
    v8f headacc = {};
    for (int ch = 0; ch < 16; ++ch) {
      __syncthreads();  // prior chunk's LDS reads done
      // Prefetch every 4th 1KB k-tile (8KB) of the next-chunk B slab (32KB/wave) into the near cache.
      // 8 prefetches at 4KB strides; per-lane addresses stride the cachelines.
      if (ch + 1 < 16) {
        const unsigned int* pf =
            wsw + ((size_t)((e * NHT + (ch + 1) * 4 + nt * 2) * NKT)) * 256 + lane * 8;
        #pragma unroll
        for (int i = 0; i < 8; ++i)
          __builtin_prefetch(pf + (size_t)i * 1024, 0, 3);   // +4KB per step
      }
      // ---------------- GEMM1: two 16x16 h tiles ----------------
      v8f acc0 = {}, acc1 = {};
      const int ht0 = ch * 4 + nt * 2;  // global hTile for ct=0
      const unsigned int* bbase =
          wsw + ((size_t)((e * NHT + ht0) * NKT)) * 256 + lane * 8;
      #pragma unroll
      for (int kt = 0; kt < NKT; ++kt) {
        AB B0, B1;
        const unsigned int* bp0 = bbase + (size_t)kt * 256;
        const unsigned int* bp1 = bp0 + (size_t)NKT * 256;  // next hTile
        B0.q[0] = *(const uint4*)bp0; B0.q[1] = *(const uint4*)(bp0 + 4);
        B1.q[0] = *(const uint4*)bp1; B1.q[1] = *(const uint4*)(bp1 + 4);
        acc0 = __builtin_amdgcn_wmma_f32_16x16x32_bf16(false, Areg[kt].v, false, B0.v, (short)0, acc0, false, false);
        acc1 = __builtin_amdgcn_wmma_f32_16x16x32_bf16(false, Areg[kt].v, false, B1.v, (short)0, acc1, false, false);
      }
      // bias + relu -> bf16 LDS.  D layout: row = rg*16 + v + hi*8, col = lane%16
      #pragma unroll
      for (int ct = 0; ct < 2; ++ct) {
        const v8f& acc = ct ? acc1 : acc0;
        int colLocal = nt * 32 + ct * 16 + c16;
        int nGlob = ch * 64 + colLocal;
        float bias = (e == 0) ? bm1[nGlob] : (e >= 2 ? b1[(size_t)(e - 2) * HH + nGlob] : 0.f);
        #pragma unroll
        for (int v = 0; v < 8; ++v) {
          float x = acc[v] + bias;
          x = x > 0.f ? x : 0.f;
          hbuf[(rg * 16 + v + hi * 8) * 64 + colLocal] = f2bf(x);
        }
      }
      __syncthreads();
      // ---------------- GEMM2: headacc += relu(h_chunk) @ Whead_chunk ----------------
      #pragma unroll
      for (int kt2 = 0; kt2 < 2; ++kt2) {
        AB A2, B2;
        const unsigned short* p = &hbuf[(rg * 16 + c16) * 64 + kt2 * 32 + hi * 16];
        A2.q[0] = *(const uint4*)(p);
        A2.q[1] = *(const uint4*)(p + 8);
        int hk = ch * 2 + kt2;
        const unsigned int* hp = hsw + ((size_t)((e * 2 + nt) * NHK + hk)) * 256 + lane * 8;
        B2.q[0] = *(const uint4*)hp; B2.q[1] = *(const uint4*)(hp + 4);
        headacc = __builtin_amdgcn_wmma_f32_16x16x32_bf16(false, A2.v, false, B2.v, (short)0, headacc, false, false);
      }
    }
    // ---------------- per-expert head epilogue ----------------
    // headacc element v, this lane: row = rg*16 + hi*8 + v, col = nt*16 + c16
    const int rbase = rg * 16 + hi * 8;
    if (e == 0) {                    // meta: logits cols 0-7, value col 8
      if (nt == 0) {
        #pragma unroll
        for (int v = 0; v < 8; ++v) {
          float x = headacc[v];
          float lx = (c16 < 8) ? x : -3.0e38f;
          float bv = lx; int bi = c16;
          for (int m = 8; m; m >>= 1) {
            float ov = __shfl_xor(bv, m, 16); int oi = __shfl_xor(bi, m, 16);
            if (ov > bv || (ov == bv && oi < bi)) { bv = ov; bi = oi; }
          }
          float ex = (c16 < 8) ? __expf(lx - bv) : 0.f;
          for (int m = 8; m; m >>= 1) ex += __shfl_xor(ex, m, 16);
          int r = rbase + v;
          if (c16 == 0) { s_meta_act[r] = (float)bi; s_meta_lp[r] = -__logf(ex); }
          if (c16 == 8) { s_meta_val[r] = x; }
        }
      }
    } else if (e == 1) {             // termination sigmoids, cols 0-7
      if (nt == 0) {
        #pragma unroll
        for (int v = 0; v < 8; ++v)
          if (c16 < 8) s_term[(rbase + v) * 8 + c16] = 1.f / (1.f + __expf(-headacc[v]));
      }
    } else {                         // option o: logits cols 0-15 (nt0), value col 16 (nt1)
      int o = e - 2;
      if (nt == 0) {
        #pragma unroll
        for (int v = 0; v < 8; ++v) {
          float x = headacc[v];
          float bv = x; int bi = c16;
          for (int m = 8; m; m >>= 1) {
            float ov = __shfl_xor(bv, m, 16); int oi = __shfl_xor(bi, m, 16);
            if (ov > bv || (ov == bv && oi < bi)) { bv = ov; bi = oi; }
          }
          float ex = __expf(x - bv);
          for (int m = 8; m; m >>= 1) ex += __shfl_xor(ex, m, 16);
          int r = rbase + v;
          if (c16 == 0) { s_oact[r * 8 + o] = (float)bi; s_olp[r * 8 + o] = -__logf(ex); }
        }
      } else {
        #pragma unroll
        for (int v = 0; v < 8; ++v)
          if (c16 == 0) s_oval[(rbase + v) * 8 + o] = headacc[v];
      }
    }
  }
  __syncthreads();
  // ---------------- final per-row select + output ----------------
  if (tid < 64) {
    int r = rowblk + tid;
    int eo = eopt[r];
    float tp = s_term[tid * 8 + eo];
    bool terminates = (dones[r] != 0) || (tp > 0.5f);
    int mact = (int)s_meta_act[tid];
    int nopt = terminates ? mact : eo;
    out[r]          = s_oact[tid * 8 + nopt];
    out[NB + r]     = s_oval[tid * 8 + nopt];
    out[2 * NB + r] = s_olp[tid * 8 + nopt];
    out[3 * NB + r] = (float)mact;
    out[4 * NB + r] = s_meta_val[tid];
    out[5 * NB + r] = s_meta_lp[tid];
    out[6 * NB + r] = tp;
  }
}

extern "C" void kernel_launch(void* const* d_in, const int* in_sizes, int n_in,
                              void* d_out, int out_size, void* d_ws, size_t ws_size,
                              hipStream_t stream) {
  const float*         obs   = (const float*)d_in[0];
  const unsigned char* dones = (const unsigned char*)d_in[1];
  const int*           eopt  = (const int*)d_in[2];
  const float* Wm1   = (const float*)d_in[3];
  const float* bm1   = (const float*)d_in[4];
  const float* Wm_pi = (const float*)d_in[5];
  const float* Wm_v  = (const float*)d_in[6];
  const float* Wt1   = (const float*)d_in[7];
  const float* Wt2   = (const float*)d_in[8];
  const float* W1    = (const float*)d_in[9];
  const float* b1    = (const float*)d_in[10];
  const float* Wpi   = (const float*)d_in[11];
  const float* Wv    = (const float*)d_in[12];

  unsigned char* ws = (unsigned char*)d_ws;
  unsigned short* obsb = (unsigned short*)ws;                         // 32 MB
  size_t off = (size_t)NB * DD * 2;
  unsigned int* wsw = (unsigned int*)(ws + off);                      // 10 MB
  off += (size_t)NE * NHT * NKT * 1024;
  unsigned int* hsw = (unsigned int*)(ws + off);                      // 640 KB

  int nObs = NB * DD;
  k_conv_obs<<<(nObs + 255) / 256, 256, 0, stream>>>(obs, obsb, nObs);
  int nW = NE * NHT * NKT * 32 * 8;
  k_swz_w<<<(nW + 255) / 256, 256, 0, stream>>>(Wm1, Wt1, W1, wsw);
  int nH = NE * 2 * NHK * 32 * 8;
  k_swz_heads<<<(nH + 255) / 256, 256, 0, stream>>>(Wm_pi, Wm_v, Wt2, Wpi, Wv, hsw);

  k_option_net<<<NB / 64, 256, 0, stream>>>(obsb, wsw, hsw, bm1, b1, dones, eopt, (float*)d_out);
}
